// RNN_Reverser_944892805643
// MI455X (gfx1250) — matrix operand order repaired and checked by reading, not compiled, not run on hardware
//
#include <hip/hip_runtime.h>

// ---------------------------------------------------------------------------
// Seq2seq Elman RNN (encoder 2-layer + decoder 2-layer + linear + log_softmax)
// MI455X / gfx1250 design:
//   * Batch-parallel decomposition: 16 persistent workgroups, each owning a
//     16-row batch slice -> the whole T=1023 recurrence runs with workgroup
//     barriers only (no global sync; encoder finals feed decoder in LDS).
//   * All matmuls via v_wmma_f32_16x16x32_bf16 (fp32 accum). Weights converted
//     once to bf16 in workspace, K padded to a multiple of 32.
//   * A fragments from LDS-staged activations, B fragments streamed from
//     WGP$/L2-resident bf16 weights each step. Weight *offsets* are laundered
//     through an empty asm per time step: blocks LICM (no VGPR hoist+spill)
//     while preserving pointer provenance (global_load, not flat_load, so the
//     weight stream uses LOADcnt only and stays off the LDS/DScnt path).
// ---------------------------------------------------------------------------

typedef unsigned short u16;
typedef __attribute__((ext_vector_type(16))) __bf16 v16bf;
typedef __attribute__((ext_vector_type(8)))  float  v8f;

constexpr int SEQ = 1024, BATCH = 256, IND = 131, INP = 160;
constexpr int HID = 256, OUTD = 131, OUTP = 144;
constexpr int T = SEQ - 1;          // 1023 steps for both encoder and decoder
constexpr int NWG = BATCH / 16;     // 16 workgroups
constexpr int NTHR = 512;           // 16 waves (wave32)

// ---- workspace layout: bf16 weights (u16 elems), then fp32 fused biases ----
constexpr int OFF_eWih0 = 0;
constexpr int OFF_eWhh0 = OFF_eWih0 + HID * INP;
constexpr int OFF_eWih1 = OFF_eWhh0 + HID * HID;
constexpr int OFF_eWhh1 = OFF_eWih1 + HID * HID;
constexpr int OFF_dWih0 = OFF_eWhh1 + HID * HID;
constexpr int OFF_dWhh0 = OFF_dWih0 + HID * INP;
constexpr int OFF_dWih1 = OFF_dWhh0 + HID * HID;
constexpr int OFF_dWhh1 = OFF_dWih1 + HID * HID;
constexpr int OFF_lin   = OFF_dWhh1 + HID * HID;      // [OUTP][HID], rows>=131 zero
constexpr int W_TOTAL   = OFF_lin + OUTP * HID;       // u16 elements
constexpr int OFF_be0 = 0, OFF_be1 = HID, OFF_bd0 = 2 * HID, OFF_bd1 = 3 * HID;
constexpr int OFF_blin = 4 * HID;                     // OUTP floats, >=131 zero

__device__ __forceinline__ u16 f2bf(float x) {       // fp32 -> bf16 RNE
  union { float f; unsigned u; } v; v.f = x;
  unsigned u = v.u;
  u += 0x7FFFu + ((u >> 16) & 1u);
  return (u16)(u >> 16);
}

// Opaque scalar: defeats LICM on weight-fragment loads (no hoist -> no spill)
// without destroying the base pointer's global address-space provenance.
__device__ __forceinline__ int opaque_i(int v) {
  asm volatile("" : "+s"(v));
  return v;
}

// A fragment: 16x32 bf16 chunk of row-major activation matrix A[16][ldk].
// Lane l: row l&15; K = k0 + ((l>>4)?8:0) + {0..7, 16..23}  (ISA 7.12.2)
__device__ __forceinline__ v16bf frag_a(const u16* A, int ldk, int k0, int lane) {
  const int r = lane & 15;
  const int k = k0 + ((lane >> 4) << 3);
  union { uint4 u[2]; v16bf v; } f;
  f.u[0] = *(const uint4*)(A + r * ldk + k);
  f.u[1] = *(const uint4*)(A + r * ldk + k + 16);
  return f.v;
}

// B fragment: 32x16 bf16 chunk of B = W^T, W row-major [N][ldk].
// Lane l: column n (= W row n); K = k0 + ((l>>4)?8:0) + {0..7, 16..23}
__device__ __forceinline__ v16bf frag_b(const u16* Wm, int ldk, int n, int k0, int lane) {
  const int k = k0 + ((lane >> 4) << 3);
  const u16* p = Wm + n * ldk + k;
  union { uint4 u[2]; v16bf v; } f;
  f.u[0] = *(const uint4*)(p);
  f.u[1] = *(const uint4*)(p + 16);
  return f.v;
}

// acc += A(16 x 32*kchunks, LDS) * W^T(tile column nlane)
__device__ __forceinline__ v8f gemm_tile(v8f acc, const u16* A, int ldkA,
                                         const u16* Wm, int ldkW,
                                         int nlane, int kchunks, int lane) {
#pragma unroll
  for (int c = 0; c < kchunks; ++c) {
    v16bf a = frag_a(A, ldkA, c * 32, lane);
    v16bf b = frag_b(Wm, ldkW, nlane, c * 32, lane);
    acc = __builtin_amdgcn_wmma_f32_16x16x32_bf16(false, a, false, b,
                                                  (short)0, acc, false, false);
  }
  return acc;
}

// ---- prep kernels -----------------------------------------------------------
__global__ void wconv_kernel(u16* __restrict__ dst, const float* __restrict__ src,
                             int Nsrc, int Ksrc, int Npad, int Kpad) {
  int total = Npad * Kpad;
  for (int i = blockIdx.x * blockDim.x + threadIdx.x; i < total;
       i += gridDim.x * blockDim.x) {
    int n = i / Kpad, k = i - n * Kpad;
    float v = (n < Nsrc && k < Ksrc) ? src[n * Ksrc + k] : 0.f;
    dst[i] = f2bf(v);
  }
}

__global__ void bias_kernel(float* __restrict__ dst, const float* __restrict__ a,
                            const float* __restrict__ b, int nsrc, int npad) {
  for (int i = blockIdx.x * blockDim.x + threadIdx.x; i < npad;
       i += gridDim.x * blockDim.x) {
    float v = 0.f;
    if (i < nsrc) { v = a[i]; if (b) v += b[i]; }
    dst[i] = v;
  }
}

// ---- persistent recurrent kernel -------------------------------------------
__global__ __launch_bounds__(NTHR)
void rnn_seq2seq_kernel(const float* __restrict__ X, const float* __restrict__ Y,
                        const u16* __restrict__ W, const float* __restrict__ BIAS,
                        float* __restrict__ out) {
  __shared__ __align__(16) u16   lds_x[16 * INP];     // step input, bf16, K-padded
  __shared__ __align__(16) u16   lds_h0[16 * HID];    // layer0 hidden (bf16)
  __shared__ __align__(16) u16   lds_h1[16 * HID];    // layer1 hidden (bf16)
  __shared__ __align__(16) float lds_logit[16 * OUTP];

  const int tid   = threadIdx.x;
  const int lane  = tid & 31;
  const int wave  = tid >> 5;                 // 0..15
  const int bbase = blockIdx.x * 16;          // batch slice
  const int n0    = wave * 16;                // output column tile
  const int nA    = n0 + (lane & 15);         // lane's output column
  const int rowg  = (lane >> 4) << 3;         // lane's row group (0 or 8)

  for (int i = tid; i < 16 * HID; i += NTHR) { lds_h0[i] = 0; lds_h1[i] = 0; }
  __syncthreads();

  // stage 16 x IND fp32 rows -> 16 x INP bf16 in LDS (zero pad K).
  // Wave w stages row w; lanes stride the padded columns (no int division).
  auto stage_x = [&](const float* xsrc) {
    const float* srow = xsrc + wave * IND;
    u16* drow = lds_x + wave * INP;
#pragma unroll
    for (int k = lane; k < INP; k += 32)
      drow[k] = f2bf((k < IND) ? srow[k] : 0.f);
    __syncthreads();
  };

  // hide HBM latency of the next step's activation row (global_prefetch_b8)
  auto prefetch_x = [&](const float* xnext) {
    if (lane < 5)  // 16 rows x 131 fp32 = 524 B/row -> 5 x 128B lines
      __builtin_prefetch((const char*)(xnext + wave * IND) + lane * 128, 0, 3);
  };

  // one RNNCell: h_dst = tanh(bias + x@Wih^T + h_src@Whh^T); A_x may be x or h
  auto cell = [&](const u16* Ax, int ldAx, int kchx, const u16* Wih,
                  const u16* Whh, u16* h_src_dst, const float* bias) {
    float bz = bias[nA];
    v8f acc; for (int j = 0; j < 8; ++j) acc[j] = bz;
    acc = gemm_tile(acc, Ax, ldAx, Wih, ldAx, nA, kchx, lane);
    acc = gemm_tile(acc, h_src_dst, HID, Whh, HID, nA, HID / 32, lane);
    __syncthreads();                          // all reads of h_src done
#pragma unroll
    for (int j = 0; j < 8; ++j)
      h_src_dst[(rowg + j) * HID + nA] = f2bf(tanhf(acc[j]));
    __syncthreads();
  };

  // ------------------------- encoder: T steps ------------------------------
  for (int t = 0; t < T; ++t) {
    stage_x(X + ((size_t)t * BATCH + bbase) * IND);
    prefetch_x(X + ((size_t)(t + 1) * BATCH + bbase) * IND);  // t+1<=T valid
    cell(lds_x,  INP, INP / 32, W + opaque_i(OFF_eWih0), W + opaque_i(OFF_eWhh0),
         lds_h0, BIAS + OFF_be0);
    cell(lds_h0, HID, HID / 32, W + opaque_i(OFF_eWih1), W + opaque_i(OFF_eWhh1),
         lds_h1, BIAS + OFF_be1);
  }
  // lds_h0/lds_h1 now hold hn0/hn1 -> decoder initial states (no handoff!)

  // ------------------------- decoder: T steps ------------------------------
  for (int t = 0; t < T; ++t) {
    const float* xsrc = (t == 0)
        ? X + ((size_t)T * BATCH + bbase) * IND          // X[-1]
        : Y + ((size_t)(t - 1) * BATCH + bbase) * IND;   // Y[t-1]
    stage_x(xsrc);
    if (t + 1 < T)
      prefetch_x(Y + ((size_t)t * BATCH + bbase) * IND);
    cell(lds_x,  INP, INP / 32, W + opaque_i(OFF_dWih0), W + opaque_i(OFF_dWhh0),
         lds_h0, BIAS + OFF_bd0);
    cell(lds_h0, HID, HID / 32, W + opaque_i(OFF_dWih1), W + opaque_i(OFF_dWhh1),
         lds_h1, BIAS + OFF_bd1);

    // logits = lin_b + h1 @ lin_W^T  (16 x OUTP, 9 column tiles)
    if (wave < OUTP / 16) {
      float bz = BIAS[OFF_blin + nA];
      v8f acc; for (int j = 0; j < 8; ++j) acc[j] = bz;
      acc = gemm_tile(acc, lds_h1, HID, W + opaque_i(OFF_lin), HID, nA,
                      HID / 32, lane);
#pragma unroll
      for (int j = 0; j < 8; ++j) lds_logit[(rowg + j) * OUTP + nA] = acc[j];
    }
    __syncthreads();

    // log_softmax over 131 logits; wave w reduces row w with lane shuffles
    {
      const float* lrow = lds_logit + wave * OUTP;
      float mx = -1e30f;
      for (int n = lane; n < OUTD; n += 32) mx = fmaxf(mx, lrow[n]);
#pragma unroll
      for (int o = 16; o >= 1; o >>= 1) mx = fmaxf(mx, __shfl_xor(mx, o, 32));
      float s = 0.f;
      for (int n = lane; n < OUTD; n += 32) s += __expf(lrow[n] - mx);
#pragma unroll
      for (int o = 16; o >= 1; o >>= 1) s += __shfl_xor(s, o, 32);
      float lse = mx + __logf(s);
      float* orow = out + ((size_t)t * BATCH + bbase + wave) * OUTD;
      for (int n = lane; n < OUTD; n += 32) orow[n] = lrow[n] - lse;
    }
    __syncthreads();                          // protect lds_logit / states
  }
}

// ---------------------------------------------------------------------------
extern "C" void kernel_launch(void* const* d_in, const int* in_sizes, int n_in,
                              void* d_out, int out_size, void* d_ws, size_t ws_size,
                              hipStream_t stream) {
  (void)in_sizes; (void)n_in; (void)out_size; (void)ws_size;
  const float* X = (const float*)d_in[0];
  const float* Y = (const float*)d_in[1];
  u16*   Wb = (u16*)d_ws;
  float* Bb = (float*)((char*)d_ws + (size_t)W_TOTAL * sizeof(u16));

  // weights -> bf16 (padded)
  wconv_kernel<<<128, 256, 0, stream>>>(Wb + OFF_eWih0, (const float*)d_in[2],  HID, IND, HID, INP);
  wconv_kernel<<<128, 256, 0, stream>>>(Wb + OFF_eWhh0, (const float*)d_in[3],  HID, HID, HID, HID);
  wconv_kernel<<<128, 256, 0, stream>>>(Wb + OFF_eWih1, (const float*)d_in[6],  HID, HID, HID, HID);
  wconv_kernel<<<128, 256, 0, stream>>>(Wb + OFF_eWhh1, (const float*)d_in[7],  HID, HID, HID, HID);
  wconv_kernel<<<128, 256, 0, stream>>>(Wb + OFF_dWih0, (const float*)d_in[10], HID, IND, HID, INP);
  wconv_kernel<<<128, 256, 0, stream>>>(Wb + OFF_dWhh0, (const float*)d_in[11], HID, HID, HID, HID);
  wconv_kernel<<<128, 256, 0, stream>>>(Wb + OFF_dWih1, (const float*)d_in[14], HID, HID, HID, HID);
  wconv_kernel<<<128, 256, 0, stream>>>(Wb + OFF_dWhh1, (const float*)d_in[15], HID, HID, HID, HID);
  wconv_kernel<<<128, 256, 0, stream>>>(Wb + OFF_lin,   (const float*)d_in[18], OUTD, HID, OUTP, HID);

  // fused biases (bih + bhh), padded lin_b
  bias_kernel<<<2, 256, 0, stream>>>(Bb + OFF_be0, (const float*)d_in[4],  (const float*)d_in[5],  HID, HID);
  bias_kernel<<<2, 256, 0, stream>>>(Bb + OFF_be1, (const float*)d_in[8],  (const float*)d_in[9],  HID, HID);
  bias_kernel<<<2, 256, 0, stream>>>(Bb + OFF_bd0, (const float*)d_in[12], (const float*)d_in[13], HID, HID);
  bias_kernel<<<2, 256, 0, stream>>>(Bb + OFF_bd1, (const float*)d_in[16], (const float*)d_in[17], HID, HID);
  bias_kernel<<<2, 256, 0, stream>>>(Bb + OFF_blin, (const float*)d_in[19], nullptr, OUTD, OUTP);

  // persistent batch-sliced recurrence: 16 WGs x 16 waves, zero global sync
  rnn_seq2seq_kernel<<<NWG, NTHR, 0, stream>>>(X, Y, Wb, Bb, (float*)d_out);
}
